// SanaLinearAttention_16209206575430
// MI455X (gfx1250) — compile-verified, matrix operand order checked by reading, not run on hardware
//
#include <hip/hip_runtime.h>
#include <hip/hip_bf16.h>

// ---------------------------------------------------------------------------
// SANA linear attention for MI455X (gfx1250, wave32, WMMA).
// B=2, S=4096, QD=INNER=2048, H=32, D=64.
// bf16 WMMA (v_wmma_f32_16x16x32_bf16) for all GEMMs, fp32 accumulation.
// Big GEMMs use double-buffered GLOBAL_LOAD_ASYNC_TO_LDS_B128 staging
// (ASYNCcnt) so HBM->LDS streaming overlaps the WMMA mainloop.
// ---------------------------------------------------------------------------

typedef __attribute__((ext_vector_type(16))) __bf16 v16bf;
typedef __attribute__((ext_vector_type(8)))  float  v8f;

#define NB   2
#define NS   4096
#define NH   32
#define ND   64
#define NQD  2048
#define NM   (NB * NS)          // 8192 rows of the big GEMMs

#define BK    64
#define LDS_K (BK + 8)          // 72 halves = 144B row stride (16B aligned)
#define TILE_HALVES (128 * LDS_K)
#define TILE_BYTES  (TILE_HALVES * 2)
#define GEMM_SMEM_BYTES (4 * TILE_BYTES)   // A0,A1,B0,B1 = 73728 B

__device__ __forceinline__ v8f wmma_bf16(v16bf a, v16bf b, v8f c) {
  return __builtin_amdgcn_wmma_f32_16x16x32_bf16(
      /*neg_a=*/false, a, /*neg_b=*/false, b,
      /*c_mod=*/(short)0, c, /*reuse_a=*/false, /*reuse_b=*/false);
}

// Load one 16x32 bf16 A/B fragment for a wave32 WMMA.
// `p` points at K=0 of this lane's row (A) / column (B); contiguous K layout.
// lanes 0-15 hold K = 0..7 and 16..23; lanes 16-31 hold K = 8..15 and 24..31.
__device__ __forceinline__ v16bf load_frag(const __bf16* p, int lane) {
  const int kb = (lane & 16) >> 1;   // 0 or 8
  v16bf f;
  ((uint4*)&f)[0] = *(const uint4*)(p + kb);
  ((uint4*)&f)[1] = *(const uint4*)(p + kb + 16);
  return f;
}

// Async 16B global -> LDS copy (per lane), tracked by ASYNCcnt.
// lds_off: LDS byte offset (dynamic-LDS-only kernels => segment starts at 0).
// gbase:   uniform 64-bit global base; goff: per-lane byte offset.
__device__ __forceinline__ void async_copy_b128(unsigned lds_off,
                                                const void* gbase,
                                                unsigned goff) {
  asm volatile("global_load_async_to_lds_b128 %0, %1, %2"
               :: "v"(lds_off), "v"(goff), "s"(gbase)
               : "memory");
}
__device__ __forceinline__ void wait_async0() {
  asm volatile("s_wait_asynccnt 0" ::: "memory");
}

// ---------------------------------------------------------------------------
// K0: fp32 -> bf16 (contiguous), 4 elements / thread
// ---------------------------------------------------------------------------
__global__ __launch_bounds__(256)
void cvt_f32_bf16_kernel(const float* __restrict__ in, __bf16* __restrict__ out,
                         size_t n) {
  size_t i = ((size_t)blockIdx.x * 256 + threadIdx.x) * 4;
  if (i + 3 < n) {
    float4 v = *(const float4*)(in + i);
    out[i + 0] = (__bf16)v.x;
    out[i + 1] = (__bf16)v.y;
    out[i + 2] = (__bf16)v.z;
    out[i + 3] = (__bf16)v.w;
  }
}

// ---------------------------------------------------------------------------
// K1: transpose + convert weight (rows x cols fp32, row-major) -> (cols x rows) bf16
// ---------------------------------------------------------------------------
__global__ __launch_bounds__(256)
void transpose_cvt_kernel(const float* __restrict__ W, __bf16* __restrict__ Wt,
                          int rows, int cols) {
  __shared__ float tile[32][33];
  const int bx = blockIdx.x * 32;            // col base
  const int by = blockIdx.y * 32;            // row base
  const int tx = threadIdx.x & 31;
  const int ty = threadIdx.x >> 5;           // 0..7
  #pragma unroll
  for (int r = ty; r < 32; r += 8)
    tile[r][tx] = W[(size_t)(by + r) * cols + bx + tx];
  __syncthreads();
  #pragma unroll
  for (int r = ty; r < 32; r += 8)
    Wt[(size_t)(bx + r) * rows + by + tx] = (__bf16)tile[tx][r];
}

// ---------------------------------------------------------------------------
// K2: fused QKV projection GEMM.
// C(M=8192, N=2048) = X_bf16 (M,K) @ Wt (N,K)^T ; K = 2048.
// blockIdx.z: 0->Q(relu) 1->K(relu) 2->V. Output bf16 in (B,H,S,D) layout.
// Block tile 128x128, K-step 64, double-buffered async LDS staging.
// 256 threads = 8 waves, each wave 4x2 WMMA tiles.
// ---------------------------------------------------------------------------
__global__ __launch_bounds__(256)
void proj_gemm_kernel(const __bf16* __restrict__ X,
                      const __bf16* __restrict__ WqT,
                      const __bf16* __restrict__ WkT,
                      const __bf16* __restrict__ WvT,
                      __bf16* __restrict__ Qo,
                      __bf16* __restrict__ Ko,
                      __bf16* __restrict__ Vo) {
  extern __shared__ __bf16 smem[];   // [A0 | A1 | B0 | B1], each 128*LDS_K halves
  const int which = blockIdx.z;
  const __bf16* Wt = (which == 0) ? WqT : (which == 1) ? WkT : WvT;
  __bf16* Out      = (which == 0) ? Qo  : (which == 1) ? Ko  : Vo;
  const bool do_relu = (which < 2);

  const int m0 = blockIdx.y * 128;
  const int n0 = blockIdx.x * 128;
  const int tid = threadIdx.x;
  const int lane = tid & 31;
  const int wave = tid >> 5;
  const int wm = (wave >> 2) * 64;  // 2 wave rows, 64 rows each
  const int wn = (wave & 3) * 32;   // 4 wave cols, 32 cols each

  v8f acc[4][2];
  #pragma unroll
  for (int i = 0; i < 4; i++)
    #pragma unroll
    for (int j = 0; j < 2; j++) acc[i][j] = v8f{};

  // async stage of one 128x64 A tile + 128x64 B tile into buffer `buf`
  auto stage = [&](int buf, int kc) {
    const __bf16* Ab = X  + (size_t)m0 * NQD + kc;
    const __bf16* Bb = Wt + (size_t)n0 * NQD + kc;
    const unsigned abase = (unsigned)(buf * TILE_BYTES);
    const unsigned bbase = (unsigned)((2 + buf) * TILE_BYTES);
    #pragma unroll
    for (int it = 0; it < 4; it++) {
      int idx = (tid + it * 256) * 8;     // 0..8191 halves
      int r = idx >> 6;
      int c = idx & 63;
      unsigned l = (unsigned)((r * LDS_K + c) * 2);
      unsigned g = (unsigned)((r * NQD + c) * 2);
      async_copy_b128(abase + l, Ab, g);
      async_copy_b128(bbase + l, Bb, g);
    }
  };

  stage(0, 0);
  wait_async0();
  __syncthreads();

  for (int kc = 0; kc < NQD; kc += BK) {
    const int cur = (kc >> 6) & 1;
    if (kc + BK < NQD) stage(cur ^ 1, kc + BK);   // prefetch next tile (async)

    const __bf16* As = smem + cur * TILE_HALVES;
    const __bf16* Bs = smem + (2 + cur) * TILE_HALVES;
    #pragma unroll
    for (int ks = 0; ks < 2; ks++) {
      v16bf af[4], bfr[2];
      #pragma unroll
      for (int i = 0; i < 4; i++)
        af[i] = load_frag(As + (wm + i * 16 + (lane & 15)) * LDS_K + ks * 32, lane);
      #pragma unroll
      for (int j = 0; j < 2; j++)
        bfr[j] = load_frag(Bs + (wn + j * 16 + (lane & 15)) * LDS_K + ks * 32, lane);
      #pragma unroll
      for (int i = 0; i < 4; i++)
        #pragma unroll
        for (int j = 0; j < 2; j++)
          acc[i][j] = wmma_bf16(af[i], bfr[j], acc[i][j]);
    }
    wait_async0();     // next buffer fully landed in LDS
    __syncthreads();   // all waves done reading current buffer
  }

  // Epilogue: C fragment: VGPR r holds M=r (lanes 0-15) / M=r+8 (lanes 16-31),
  // N = lane&15.  (m,n) -> Out[((b*H + h)*S + s)*D + d]
  #pragma unroll
  for (int i = 0; i < 4; i++) {
    #pragma unroll
    for (int j = 0; j < 2; j++) {
      #pragma unroll
      for (int r = 0; r < 8; r++) {
        int m = m0 + wm + i * 16 + r + ((lane & 16) >> 1);
        int n = n0 + wn + j * 16 + (lane & 15);
        float v = acc[i][j][r];
        if (do_relu) v = v > 0.f ? v : 0.f;
        int b = m >> 12;            // / 4096
        int s = m & (NS - 1);
        int h = n >> 6;
        int d = n & 63;
        Out[(((size_t)(b * NH + h) * NS) + s) * ND + d] = (__bf16)v;
      }
    }
  }
}

// ---------------------------------------------------------------------------
// K3: key_sum[bh][d] = sum_s K[bh][s][d]
// ---------------------------------------------------------------------------
__global__ __launch_bounds__(256)
void ksum_kernel(const __bf16* __restrict__ Kmat, float* __restrict__ ksum) {
  __shared__ float red[256];
  const int bh = blockIdx.x;
  const int d = threadIdx.x & 63;
  const int si = threadIdx.x >> 6;
  const __bf16* Kb = Kmat + (size_t)bh * NS * ND;
  float s = 0.f;
  for (int sidx = si; sidx < NS; sidx += 4)
    s += (float)Kb[(size_t)sidx * ND + d];
  red[threadIdx.x] = s;
  __syncthreads();
  if (threadIdx.x < 64)
    ksum[bh * ND + d] = red[d] + red[64 + d] + red[128 + d] + red[192 + d];
}

// ---------------------------------------------------------------------------
// K4: kv[bh] = K^T @ V  (64x64, reduction over S=4096) via WMMA.
// One block per (b,h). 8 waves; wave w -> row tile (w>>1), col tiles 2*(w&1)+{0,1}.
// ---------------------------------------------------------------------------
__global__ __launch_bounds__(256)
void kv_kernel(const __bf16* __restrict__ Kmat, const __bf16* __restrict__ Vmat,
               float* __restrict__ kv) {
  __shared__ __bf16 Ks[64][LDS_K];   // [d][s_local]  (transposed stage)
  __shared__ __bf16 Vs[64][LDS_K];   // [e][s_local]
  const int bh = blockIdx.x;
  const int tid = threadIdx.x;
  const int lane = tid & 31;
  const int wave = tid >> 5;
  const int tm = (wave >> 1) * 16;   // 0,16,32,48
  const int tn = (wave & 1) * 32;    // 0,32
  v8f acc[2];
  acc[0] = v8f{};
  acc[1] = v8f{};
  const __bf16* Kb = Kmat + (size_t)bh * NS * ND;
  const __bf16* Vb = Vmat + (size_t)bh * NS * ND;

  for (int s0 = 0; s0 < NS; s0 += 64) {
    #pragma unroll
    for (int it = 0; it < 2; it++) {
      int r  = (tid >> 3) + it * 32;   // s local 0..63
      int c0 = (tid & 7) * 8;          // d/e base
      const __bf16* kp = Kb + (size_t)(s0 + r) * ND + c0;
      const __bf16* vp = Vb + (size_t)(s0 + r) * ND + c0;
      #pragma unroll
      for (int j = 0; j < 8; j++) {
        Ks[c0 + j][r] = kp[j];
        Vs[c0 + j][r] = vp[j];
      }
    }
    __syncthreads();
    #pragma unroll
    for (int ks = 0; ks < 2; ks++) {
      v16bf af  = load_frag(&Ks[tm + (lane & 15)][ks * 32], lane);
      v16bf bf0 = load_frag(&Vs[tn + (lane & 15)][ks * 32], lane);
      v16bf bf1 = load_frag(&Vs[tn + 16 + (lane & 15)][ks * 32], lane);
      acc[0] = wmma_bf16(af, bf0, acc[0]);
      acc[1] = wmma_bf16(af, bf1, acc[1]);
    }
    __syncthreads();
  }
  #pragma unroll
  for (int j = 0; j < 2; j++)
    #pragma unroll
    for (int r = 0; r < 8; r++) {
      int d = tm + r + ((lane & 16) >> 1);
      int e = tn + j * 16 + (lane & 15);
      kv[(size_t)bh * (ND * ND) + d * ND + e] = acc[j][r];
    }
}

// ---------------------------------------------------------------------------
// K5: normalizer[bh][s] = max(1e-6, dot(Q[bh][s][:], key_sum[bh][:]))
// ---------------------------------------------------------------------------
__global__ __launch_bounds__(256)
void norm_kernel(const __bf16* __restrict__ Q, const float* __restrict__ ksum,
                 float* __restrict__ nrm) {
  const int idx = blockIdx.x * 256 + threadIdx.x;    // bh*S + s
  const int bh = idx >> 12;
  const __bf16* q = Q + (size_t)idx * ND;
  const float* ks = ksum + bh * ND;
  float acc = 0.f;
  #pragma unroll
  for (int d = 0; d < ND; d++) acc += (float)q[d] * ks[d];
  nrm[idx] = fmaxf(acc, 1e-6f);
}

// ---------------------------------------------------------------------------
// K6: attn = (Q @ kv) / normalizer, bf16, layout (B, S, H*D).
// Grid (BH, S/128). 8 waves, each wave: 16 rows x 4 col-tiles. K-dim = 64.
// ---------------------------------------------------------------------------
__global__ __launch_bounds__(256)
void qkv_kernel(const __bf16* __restrict__ Q, const float* __restrict__ kv,
                const float* __restrict__ nrm, __bf16* __restrict__ attn) {
  __shared__ __bf16 kvT[64][LDS_K];   // [e][d]
  const int bh = blockIdx.x;
  const int b = bh >> 5, h = bh & 31;
  const int s0 = blockIdx.y * 128;
  const int tid = threadIdx.x;
  const int lane = tid & 31;
  const int wave = tid >> 5;

  const float* kvb = kv + (size_t)bh * (ND * ND);
  for (int i = tid; i < ND * ND; i += 256) {
    int d = i >> 6, e = i & 63;
    kvT[e][d] = (__bf16)kvb[i];
  }
  __syncthreads();

  const __bf16* Qb = Q + ((size_t)bh * NS + s0 + wave * 16) * ND;
  v8f acc[4];
  #pragma unroll
  for (int j = 0; j < 4; j++) acc[j] = v8f{};

  #pragma unroll
  for (int kc = 0; kc < 2; kc++) {
    v16bf af = load_frag(Qb + (size_t)(lane & 15) * ND + kc * 32, lane);
    #pragma unroll
    for (int j = 0; j < 4; j++) {
      v16bf bfr = load_frag(&kvT[j * 16 + (lane & 15)][kc * 32], lane);
      acc[j] = wmma_bf16(af, bfr, acc[j]);
    }
  }

  const float* nb = nrm + (size_t)bh * NS + s0 + wave * 16;
  #pragma unroll
  for (int j = 0; j < 4; j++)
    #pragma unroll
    for (int r = 0; r < 8; r++) {
      int ml = r + ((lane & 16) >> 1);
      int s  = s0 + wave * 16 + ml;
      int e  = j * 16 + (lane & 15);
      float v = acc[j][r] / nb[ml];
      attn[((size_t)(b * NS + s)) * NQD + h * ND + e] = (__bf16)v;
    }
}

// ---------------------------------------------------------------------------
// K7: out(M=8192, N=2048) = attn_bf16 @ WoT^T + bo, fp32 output.
// Same double-buffered async-staged mainloop as K2.
// ---------------------------------------------------------------------------
__global__ __launch_bounds__(256)
void out_gemm_kernel(const __bf16* __restrict__ A, const __bf16* __restrict__ WoT,
                     const float* __restrict__ bo, float* __restrict__ out) {
  extern __shared__ __bf16 smem[];
  const int m0 = blockIdx.y * 128;
  const int n0 = blockIdx.x * 128;
  const int tid = threadIdx.x;
  const int lane = tid & 31;
  const int wave = tid >> 5;
  const int wm = (wave >> 2) * 64;
  const int wn = (wave & 3) * 32;

  v8f acc[4][2];
  #pragma unroll
  for (int i = 0; i < 4; i++)
    #pragma unroll
    for (int j = 0; j < 2; j++) acc[i][j] = v8f{};

  auto stage = [&](int buf, int kc) {
    const __bf16* Ab = A   + (size_t)m0 * NQD + kc;
    const __bf16* Bb = WoT + (size_t)n0 * NQD + kc;
    const unsigned abase = (unsigned)(buf * TILE_BYTES);
    const unsigned bbase = (unsigned)((2 + buf) * TILE_BYTES);
    #pragma unroll
    for (int it = 0; it < 4; it++) {
      int idx = (tid + it * 256) * 8;
      int r = idx >> 6;
      int c = idx & 63;
      unsigned l = (unsigned)((r * LDS_K + c) * 2);
      unsigned g = (unsigned)((r * NQD + c) * 2);
      async_copy_b128(abase + l, Ab, g);
      async_copy_b128(bbase + l, Bb, g);
    }
  };

  stage(0, 0);
  wait_async0();
  __syncthreads();

  for (int kc = 0; kc < NQD; kc += BK) {
    const int cur = (kc >> 6) & 1;
    if (kc + BK < NQD) stage(cur ^ 1, kc + BK);

    const __bf16* As = smem + cur * TILE_HALVES;
    const __bf16* Bs = smem + (2 + cur) * TILE_HALVES;
    #pragma unroll
    for (int ks = 0; ks < 2; ks++) {
      v16bf af[4], bfr[2];
      #pragma unroll
      for (int i = 0; i < 4; i++)
        af[i] = load_frag(As + (wm + i * 16 + (lane & 15)) * LDS_K + ks * 32, lane);
      #pragma unroll
      for (int j = 0; j < 2; j++)
        bfr[j] = load_frag(Bs + (wn + j * 16 + (lane & 15)) * LDS_K + ks * 32, lane);
      #pragma unroll
      for (int i = 0; i < 4; i++)
        #pragma unroll
        for (int j = 0; j < 2; j++)
          acc[i][j] = wmma_bf16(af[i], bfr[j], acc[i][j]);
    }
    wait_async0();
    __syncthreads();
  }

  #pragma unroll
  for (int i = 0; i < 4; i++) {
    #pragma unroll
    for (int j = 0; j < 2; j++) {
      int n = n0 + wn + j * 16 + (lane & 15);
      float bias = bo[n];
      #pragma unroll
      for (int r = 0; r < 8; r++) {
        int m = m0 + wm + i * 16 + r + ((lane & 16) >> 1);
        out[(size_t)m * NQD + n] = acc[i][j][r] + bias;
      }
    }
  }
}

// ---------------------------------------------------------------------------
// Host launcher
// ---------------------------------------------------------------------------
extern "C" void kernel_launch(void* const* d_in, const int* in_sizes, int n_in,
                              void* d_out, int out_size, void* d_ws, size_t ws_size,
                              hipStream_t stream) {
  const float* X_f  = (const float*)d_in[0];   // (B,S,QD)
  const float* Wq_f = (const float*)d_in[1];   // (QD, INNER)
  const float* Wk_f = (const float*)d_in[2];
  const float* Wv_f = (const float*)d_in[3];
  const float* Wo_f = (const float*)d_in[4];   // (INNER, QD)
  const float* bo_f = (const float*)d_in[5];   // (QD,)
  float* out = (float*)d_out;

  // Workspace carve-up (~204 MB total)
  char* w = (char*)d_ws;
  const size_t XB_BYTES = (size_t)NM * NQD * 2;   // 33.5 MB
  const size_t WT_BYTES = (size_t)NQD * NQD * 2;  // 8.4 MB each
  __bf16* Xb   = (__bf16*)w; w += XB_BYTES;
  __bf16* WqT  = (__bf16*)w; w += WT_BYTES;
  __bf16* WkT  = (__bf16*)w; w += WT_BYTES;
  __bf16* WvT  = (__bf16*)w; w += WT_BYTES;
  __bf16* WoT  = (__bf16*)w; w += WT_BYTES;
  __bf16* Qb   = (__bf16*)w; w += XB_BYTES;       // (B,H,S,D) bf16
  __bf16* Kb   = (__bf16*)w; w += XB_BYTES;
  __bf16* Vb   = (__bf16*)w; w += XB_BYTES;
  __bf16* attn = (__bf16*)w; w += XB_BYTES;       // (B,S,INNER) bf16
  float*  kv   = (float*)w;  w += (size_t)NB * NH * ND * ND * 4;
  float*  ksum = (float*)w;  w += (size_t)NB * NH * ND * 4;
  float*  nrm  = (float*)w;  w += (size_t)NB * NH * NS * 4;

  // K0: X -> bf16
  {
    size_t n = (size_t)NM * NQD;
    cvt_f32_bf16_kernel<<<dim3((unsigned)(n / (4 * 256))), dim3(256), 0, stream>>>(X_f, Xb, n);
  }
  // K1: transpose+convert the four weights to (N,K) bf16
  {
    dim3 g(NQD / 32, NQD / 32);
    transpose_cvt_kernel<<<g, 256, 0, stream>>>(Wq_f, WqT, NQD, NQD);
    transpose_cvt_kernel<<<g, 256, 0, stream>>>(Wk_f, WkT, NQD, NQD);
    transpose_cvt_kernel<<<g, 256, 0, stream>>>(Wv_f, WvT, NQD, NQD);
    transpose_cvt_kernel<<<g, 256, 0, stream>>>(Wo_f, WoT, NQD, NQD);
  }
  // K2: fused QKV projection GEMM (z = which weight), async double-buffered
  proj_gemm_kernel<<<dim3(NQD / 128, NM / 128, 3), 256, GEMM_SMEM_BYTES, stream>>>(
      Xb, WqT, WkT, WvT, Qb, Kb, Vb);
  // K3: key_sum
  ksum_kernel<<<dim3(NB * NH), 256, 0, stream>>>(Kb, ksum);
  // K4: kv = K^T @ V per head
  kv_kernel<<<dim3(NB * NH), 256, 0, stream>>>(Kb, Vb, kv);
  // K5: normalizer
  norm_kernel<<<dim3((NB * NH * NS) / 256), 256, 0, stream>>>(Qb, ksum, nrm);
  // K6: qkv / normalizer -> attn bf16
  qkv_kernel<<<dim3(NB * NH, NS / 128), 256, 0, stream>>>(Qb, kv, nrm, attn);
  // K7: out = attn @ Wo + bo, async double-buffered
  out_gemm_kernel<<<dim3(NQD / 128, NM / 128), 256, GEMM_SMEM_BYTES, stream>>>(
      attn, WoT, bo_f, out);
}
